// QueryAndAttend_41489384080046
// MI455X (gfx1250) — matrix operand order repaired and checked by reading, not compiled, not run on hardware
//
#include <hip/hip_runtime.h>

// ---------------------------------------------------------------------------
// QueryAndAttend fused pipeline for MI455X (gfx1250, wave32, WMMA bf16 + TDM)
//
//  K1: kv  = w_kv @ x          (bf16 WMMA, f32 acc)       -> ws
//  K2: sim = (q*scale) @ k     (VALU, memory-bound)       -> ws
//  K3: softmax(3x3)+attn@v  fused with  out = w_out @ y.
//      w_out staged into LDS by the Tensor Data Mover (bank-padded pitch via
//      D# pad fields) overlapped with the softmax phase; GEMM phase is pure
//      ds_load_b128 + v_wmma_f32_16x16x32_bf16.
// ---------------------------------------------------------------------------

typedef __attribute__((ext_vector_type(16))) __bf16 v16bf;
typedef __attribute__((ext_vector_type(8)))  float  v8f;
typedef __attribute__((ext_vector_type(4)))  unsigned int v4u;
typedef __attribute__((ext_vector_type(8)))  int    v8i;
typedef __attribute__((ext_vector_type(4)))  int    v4i;

union Frag16 { v16bf v; uint4 q[2]; };
union Pack4  { __bf16 h[4]; uint2 u; };

#define HW     16384      // 128*128
#define IMW    128
#define CIN    256
#define NH     8
#define DH     32
#define YPITCH 264        // 256 + 8 halves (528 B rows -> bank stride 4/lane)
#define SMEM_BYTES ((128 + 256) * YPITCH * 2)   // Y tile + padded w_out

static __device__ __forceinline__ v8f wmma_bf16(v16bf a, v16bf b, v8f c) {
    return __builtin_amdgcn_wmma_f32_16x16x32_bf16(
        false, a, false, b, (short)0, c, false, false);
}

// ---------------------------------------------------------------- kernel 0 --
__global__ void cvt_weights(const float* __restrict__ wkv,
                            const float* __restrict__ wout,
                            __bf16* __restrict__ wkv_b,
                            __bf16* __restrict__ wout_b) {
    int i = blockIdx.x * 256 + threadIdx.x;
    if (i < 64 * 256)  wkv_b[i]  = (__bf16)wkv[i];
    if (i < 256 * 256) wout_b[i] = (__bf16)wout[i];
}

// ---------------------------------------------------------------- kernel 1 --
// kv[b][0:64][p] = w_kv @ x[b][:, p].  One wave: 64 rows x 16 pixels.
__global__ __launch_bounds__(256) void gemm_kv(const float* __restrict__ x,
                                               const __bf16* __restrict__ wkv,
                                               float* __restrict__ kv) {
    const int lane = threadIdx.x & 31;
    const int gw   = blockIdx.x * 8 + (threadIdx.x >> 5);
    const int bidx = gw >> 10;
    const int p0   = (gw & 1023) * 16;
    const int ln   = lane & 15, lh = lane >> 4;

    const float* xb = x + (size_t)bidx * CIN * HW;

    v8f acc[4];
    const v8f vz = {0.f,0.f,0.f,0.f,0.f,0.f,0.f,0.f};
#pragma unroll
    for (int mt = 0; mt < 4; ++mt) acc[mt] = vz;

    for (int k0 = 0; k0 < CIN; k0 += 32) {
        v16bf bfrag;                      // lanes 0-15: K=k0.., lanes 16-31: +16
        const float* xp = xb + (size_t)(k0 + lh * 16) * HW + p0 + ln;
#pragma unroll
        for (int i = 0; i < 16; ++i) bfrag[i] = (__bf16)xp[(size_t)i * HW];

#pragma unroll
        for (int mt = 0; mt < 4; ++mt) {
            Frag16 a;                     // row=mt*16+ln; 16 contiguous K (+16 for lanes 16-31)
            const __bf16* wp = wkv + (mt * 16 + ln) * CIN + k0 + lh * 16;
            a.q[0] = *(const uint4*)(wp);
            a.q[1] = *(const uint4*)(wp + 8);
            acc[mt] = wmma_bf16(a.v, bfrag, acc[mt]);
        }
    }

    float* kvb = kv + (size_t)bidx * 64 * HW;
#pragma unroll
    for (int mt = 0; mt < 4; ++mt)
#pragma unroll
        for (int r = 0; r < 8; ++r)
            kvb[(size_t)(mt * 16 + r + lh * 8) * HW + p0 + ln] = acc[mt][r];
}

// ---------------------------------------------------------------- kernel 2 --
__global__ void sim_kernel(const float* __restrict__ kv,
                           const float* __restrict__ queries,
                           float* __restrict__ sim) {
    const int g    = blockIdx.x * 256 + threadIdx.x;
    const int bidx = g >> 14;
    const int p    = g & (HW - 1);
    const float scale = 0.17677669529663687f;   // 32^-0.5

    const float* kb = kv + (size_t)bidx * 64 * HW + p;
    float kvec[DH];
#pragma unroll
    for (int d = 0; d < DH; ++d) kvec[d] = kb[(size_t)d * HW];

#pragma unroll
    for (int h = 0; h < NH; ++h) {
        float s = 0.f;
#pragma unroll
        for (int d = 0; d < DH; ++d) s += queries[h * DH + d] * kvec[d];
        sim[((size_t)bidx * NH + h) * HW + p] = s * scale;
    }
}

// ---------------------------------------------------------------- kernel 3 --
// One workgroup = one image row (128 px) of one batch.
__global__ __launch_bounds__(256, 1) void attn_out(const float* __restrict__ kv,
                                                   const float* __restrict__ sim,
                                                   const float* __restrict__ bias,
                                                   const __bf16* __restrict__ wout,
                                                   float* __restrict__ out) {
    extern __shared__ __align__(16) char smem[];
    __bf16* Yl = (__bf16*)smem;                           // [128][YPITCH]
    __bf16* Wl = (__bf16*)(smem + 128 * YPITCH * 2);      // [256][YPITCH]

    const int bidx = blockIdx.x >> 7;
    const int row  = blockIdx.x & 127;

    // ---- TDM: stage w_out (256x256 bf16) into LDS with 16B row padding ----
    if (threadIdx.x < 32) {
        const unsigned lds_off = (unsigned)(size_t)(void*)Wl;
        const unsigned long long ga = (unsigned long long)(size_t)wout;
        // group0: count=1 | lds_addr | global_addr[56:0] | type=2
        v4u g0 = { 1u, lds_off, (unsigned)ga,
                   (unsigned)((ga >> 32) & 0x1FFFFFFu) | 0x80000000u };
        // group1: data_size=2B, pad_enable, pad_interval=512B, pad_amount=16B,
        //         tensor_dim0=256, tensor_dim1=256, tile 256x256, stride0=256
        v8i g1 = { (int)((1u << 16) | (1u << 20) | (6u << 22) | (3u << 25)),
                   (int)(256u << 16),   // tensor_dim0 -> bits[79:48]
                   (int)(256u << 16),   // tensor_dim1 -> bits[111:80]
                   (int)(256u << 16),   // tile_dim0   -> bits[127:112]
                   256,                 // tile_dim1   -> bits[143:128]
                   256,                 // tensor_dim0_stride -> bits[207:160]
                   0, 0 };
        v4i gz = {0, 0, 0, 0};
#if __clang_major__ >= 23
        v8i gz8 = {0, 0, 0, 0, 0, 0, 0, 0};
        __builtin_amdgcn_tensor_load_to_lds(g0, g1, gz, gz, gz8, 0);
#else
        __builtin_amdgcn_tensor_load_to_lds(g0, g1, gz, gz, 0);
#endif
    }

    // ---- phase 1: 3x3 softmax + attn @ v -> Yl[px][channel] (bf16) --------
    {
        const int t  = threadIdx.x;
        const int px = t & 127;
        const int hb = (t >> 7) * 4;                  // heads hb..hb+3
        const float* simb = sim + (size_t)bidx * NH * HW;
        const float* vb   = kv + ((size_t)bidx * 64 + DH) * HW;

        float attn[4][9];
#pragma unroll
        for (int hh = 0; hh < 4; ++hh) {
            const int h = hb + hh;
            float mx = -3.4e38f;
#pragma unroll
            for (int j = 0; j < 9; ++j) {
                const int yy = row + (j / 3) - 1;
                const int xx = px  + (j % 3) - 1;
                float s = -3.0e38f;
                if (yy >= 0 && yy < IMW && xx >= 0 && xx < IMW)
                    s = simb[(size_t)h * HW + yy * IMW + xx] + bias[h * 9 + j];
                attn[hh][j] = s;
                mx = fmaxf(mx, s);
            }
            float sum = 0.f;
#pragma unroll
            for (int j = 0; j < 9; ++j) {
                const float e = __expf(attn[hh][j] - mx);
                attn[hh][j] = e;
                sum += e;
            }
            const float inv = 1.0f / sum;
#pragma unroll
            for (int j = 0; j < 9; ++j) attn[hh][j] *= inv;
        }

#pragma unroll
        for (int d0 = 0; d0 < DH; d0 += 4) {
            float a4[4][4];
#pragma unroll
            for (int hh = 0; hh < 4; ++hh)
#pragma unroll
                for (int s = 0; s < 4; ++s) a4[hh][s] = 0.f;

#pragma unroll
            for (int j = 0; j < 9; ++j) {
                const int yy = row + (j / 3) - 1;
                const int xx = px  + (j % 3) - 1;
                float vv[4] = {0.f, 0.f, 0.f, 0.f};
                if (yy >= 0 && yy < IMW && xx >= 0 && xx < IMW) {
                    const float* vp = vb + (size_t)d0 * HW + yy * IMW + xx;
#pragma unroll
                    for (int s = 0; s < 4; ++s) vv[s] = vp[(size_t)s * HW];
                }
#pragma unroll
                for (int hh = 0; hh < 4; ++hh)
#pragma unroll
                    for (int s = 0; s < 4; ++s)
                        a4[hh][s] += attn[hh][j] * vv[s];
            }
#pragma unroll
            for (int hh = 0; hh < 4; ++hh) {
                Pack4 pk;
#pragma unroll
                for (int s = 0; s < 4; ++s) pk.h[s] = (__bf16)a4[hh][s];
                *(uint2*)&Yl[px * YPITCH + (hb + hh) * DH + d0] = pk.u;
            }
        }
    }

    // TDM completion (only wave 0 has TENSORcnt pending), then block barrier.
    if (threadIdx.x < 32) {
#if __has_builtin(__builtin_amdgcn_s_wait_tensorcnt)
        __builtin_amdgcn_s_wait_tensorcnt(0);
#else
        asm volatile("s_wait_tensorcnt 0x0" ::: "memory");
#endif
    }
    __syncthreads();

    // ---- phase 2: out = w_out @ Y, all operands from LDS -------------------
    {
        const int lane = threadIdx.x & 31;
        const int wv   = threadIdx.x >> 5;            // 8 waves -> 8 px columns
        const int ln   = lane & 15, lh = lane >> 4;
        const int pc   = wv * 16;

        v8f acc[16];
        const v8f vz = {0.f,0.f,0.f,0.f,0.f,0.f,0.f,0.f};
#pragma unroll
        for (int mt = 0; mt < 16; ++mt) acc[mt] = vz;

        for (int k0 = 0; k0 < CIN; k0 += 32) {
            Frag16 bf;                                // 16 contiguous K halves
            const __bf16* yp = Yl + (pc + ln) * YPITCH + k0 + lh * 16;
            bf.q[0] = *(const uint4*)(yp);
            bf.q[1] = *(const uint4*)(yp + 8);

#pragma unroll
            for (int mt = 0; mt < 16; ++mt) {
                Frag16 a;
                const __bf16* wp = Wl + (mt * 16 + ln) * YPITCH + k0 + lh * 16;
                a.q[0] = *(const uint4*)(wp);
                a.q[1] = *(const uint4*)(wp + 8);
                acc[mt] = wmma_bf16(a.v, bf.v, acc[mt]);
            }
        }

        float* ob = out + (size_t)bidx * CIN * HW + row * IMW;
#pragma unroll
        for (int mt = 0; mt < 16; ++mt)
#pragma unroll
            for (int r = 0; r < 8; ++r)
                ob[(size_t)(mt * 16 + r + lh * 8) * HW + pc + ln] = acc[mt][r];
    }
}

// ---------------------------------------------------------------- launcher --
extern "C" void kernel_launch(void* const* d_in, const int* in_sizes, int n_in,
                              void* d_out, int out_size, void* d_ws, size_t ws_size,
                              hipStream_t stream) {
    const float* x       = (const float*)d_in[0];
    const float* queries = (const float*)d_in[1];
    const float* bias    = (const float*)d_in[2];
    const float* w_kv    = (const float*)d_in[3];
    const float* w_out   = (const float*)d_in[4];
    float* out = (float*)d_out;

    char* ws = (char*)d_ws;
    float*  kvbuf  = (float*)ws;                                   // 64 MiB
    float*  simbuf = (float*)(ws + (size_t)67108864);              // 8 MiB
    __bf16* wkv_b  = (__bf16*)(ws + (size_t)67108864 + 8388608);   // 32 KiB
    __bf16* wout_b = (__bf16*)(ws + (size_t)67108864 + 8388608 + 32768);

    cvt_weights<<<256, 256, 0, stream>>>(w_kv, w_out, wkv_b, wout_b);
    gemm_kv   <<<2048, 256, 0, stream>>>(x, wkv_b, kvbuf);
    sim_kernel<<<1024, 256, 0, stream>>>(kvbuf, queries, simbuf);
    attn_out  <<<2048, 256, SMEM_BYTES, stream>>>(kvbuf, simbuf, bias, wout_b, out);
}
